// MGRUF_55241869361388
// MI455X (gfx1250) — compile-verified, matrix operand order checked by reading, not run on hardware
//
#include <hip/hip_runtime.h>
#include <hip/hip_bf16.h>

// MGRU-F (fractional-order GRU) for MI455X / gfx1250.
// T=256 B=64 I=512 H=1024 O=512 K=8.
// Latency-bound 256-step recurrence -> bf16x3 split-precision WMMA
// (v_wmma_f32_16x16x32_bf16); weights pre-converted once per launch into
// fragment-major bf16 hi/lo; B fragments software-pipelined one k-slab
// ahead with a 2x-unrolled ping-pong (no register rotation copies);
// A staging uses float4 loads + packed bf16 ds_store_b64.

#define T_DIM 256
#define B_DIM 64
#define I_DIM 512
#define H_DIM 1024
#define O_DIM 512
#define K_DIM 8
#define G3    (3 * H_DIM)

typedef __attribute__((ext_vector_type(16))) __bf16 v16bf;
typedef __attribute__((ext_vector_type(8)))  float  v8f;

union FragU { uint4 q[2]; v16bf v; };
union PackU { __bf16 h[16]; uint4 q[2]; };

// Fragment lane mapping per CDNA5 ISA 7.12.2 (16-bit A, 16x32):
//   lane<16 : row M = lane,     K in {0..7} u {16..23}
//   lane>=16: row M = lane-16,  K in {8..15} u {24..31}
// B (32x16) assumed dual: lane -> column N, same K striping.

__device__ __forceinline__ v16bf load_frag(const __bf16* row, int half) {
    FragU f;
    f.q[0] = *reinterpret_cast<const uint4*>(row + half * 8);
    f.q[1] = *reinterpret_cast<const uint4*>(row + 16 + half * 8);
    return f.v;
}

__device__ __forceinline__ v16bf load_gfrag(const __bf16* p) {
    FragU f;
    f.q[0] = *reinterpret_cast<const uint4*>(p);
    f.q[1] = *reinterpret_cast<const uint4*>(p + 8);
    return f.v;
}

__device__ __forceinline__ const __bf16* fragptr(const __bf16* F, int jb, int g,
                                                 int KT, int kt, int lane) {
    return F + ((((size_t)jb * 3 + g) * KT + kt) * 32 + lane) * 16;
}

// fp32-accurate product via bf16 split: a*b ~ alo*bhi + ahi*blo + ahi*bhi
__device__ __forceinline__ v8f wmma3(v16bf ahi, v16bf alo, v16bf bhi, v16bf blo, v8f acc) {
    acc = __builtin_amdgcn_wmma_f32_16x16x32_bf16(false, alo, false, bhi, (short)0, acc, false, false);
    acc = __builtin_amdgcn_wmma_f32_16x16x32_bf16(false, ahi, false, blo, (short)0, acc, false, false);
    acc = __builtin_amdgcn_wmma_f32_16x16x32_bf16(false, ahi, false, bhi, (short)0, acc, false, false);
    return acc;
}

__device__ __forceinline__ float sigmf(float x) { return 1.0f / (1.0f + expf(-x)); }

// Six B fragments for one k-slab (3 gates x hi/lo).
struct BF6 { v16bf h0, l0, h1, l1, h2, l2; };

__device__ __forceinline__ BF6 load_b6(const __bf16* __restrict__ Fhi,
                                       const __bf16* __restrict__ Flo,
                                       int jb, int KT, int kt, int lane) {
    BF6 b;
    b.h0 = load_gfrag(fragptr(Fhi, jb, 0, KT, kt, lane));
    b.l0 = load_gfrag(fragptr(Flo, jb, 0, KT, kt, lane));
    b.h1 = load_gfrag(fragptr(Fhi, jb, 1, KT, kt, lane));
    b.l1 = load_gfrag(fragptr(Flo, jb, 1, KT, kt, lane));
    b.h2 = load_gfrag(fragptr(Fhi, jb, 2, KT, kt, lane));
    b.l2 = load_gfrag(fragptr(Flo, jb, 2, KT, kt, lane));
    return b;
}

// Stage a 64x32 f32 A-slab into bf16 hi/lo LDS: float4 loads, b64 stores.
__device__ __forceinline__ void stage_A(const float* __restrict__ src, int lda, int k0,
                                        __bf16 (&hi)[64][32], __bf16 (&lo)[64][32], int tid) {
    for (int idx = tid; idx < 64 * 8; idx += 128) {   // 8 float4-quads per row
        int row = idx >> 3, kq = (idx & 7) * 4;
        float4 v = *reinterpret_cast<const float4*>(&src[row * lda + k0 + kq]);
        union { __bf16 h[4]; uint2 u; } ph, pl;
        ph.h[0] = (__bf16)v.x; ph.h[1] = (__bf16)v.y;
        ph.h[2] = (__bf16)v.z; ph.h[3] = (__bf16)v.w;
        pl.h[0] = (__bf16)(v.x - (float)ph.h[0]);
        pl.h[1] = (__bf16)(v.y - (float)ph.h[1]);
        pl.h[2] = (__bf16)(v.z - (float)ph.h[2]);
        pl.h[3] = (__bf16)(v.w - (float)ph.h[3]);
        *reinterpret_cast<uint2*>(&hi[row][kq]) = ph.u;
        *reinterpret_cast<uint2*>(&lo[row][kq]) = pl.u;
    }
}

// One GEMM phase: acc{0,1,2} += A(64 x 32*KT) @ Wg (3 gate column-slices).
// 2x-unrolled ping-pong: fragments f0/f1 and LDS buffers 0/1 alternate roles,
// so the software pipeline has no loop-carried register copies.
__device__ __forceinline__ void gemm_phase(
    const float* __restrict__ A, int lda,
    const __bf16* __restrict__ Fhi, const __bf16* __restrict__ Flo,
    int KT, int jb, int tid, int lane, int m0, int lr, int half,
    __bf16 (&AsHi)[2][64][32], __bf16 (&AsLo)[2][64][32],
    v8f& acc0, v8f& acc1, v8f& acc2) {
    stage_A(A, lda, 0, AsHi[0], AsLo[0], tid);
    BF6 f0 = load_b6(Fhi, Flo, jb, KT, 0, lane);
    BF6 f1 = {};
    __syncthreads();
    for (int kt = 0; kt < KT; kt += 2) {   // KT is even
        {   // even slab kt: consume f0 / buffer 0; prefetch kt+1 into f1 / buffer 1
            v16bf ahi = load_frag(&AsHi[0][m0 + lr][0], half);
            v16bf alo = load_frag(&AsLo[0][m0 + lr][0], half);
            f1 = load_b6(Fhi, Flo, jb, KT, kt + 1, lane);
            stage_A(A, lda, (kt + 1) * 32, AsHi[1], AsLo[1], tid);
            acc0 = wmma3(ahi, alo, f0.h0, f0.l0, acc0);
            acc1 = wmma3(ahi, alo, f0.h1, f0.l1, acc1);
            acc2 = wmma3(ahi, alo, f0.h2, f0.l2, acc2);
            __syncthreads();
        }
        {   // odd slab kt+1: consume f1 / buffer 1; prefetch kt+2 into f0 / buffer 0
            v16bf ahi = load_frag(&AsHi[1][m0 + lr][0], half);
            v16bf alo = load_frag(&AsLo[1][m0 + lr][0], half);
            if (kt + 2 < KT) {
                f0 = load_b6(Fhi, Flo, jb, KT, kt + 2, lane);
                stage_A(A, lda, (kt + 2) * 32, AsHi[0], AsLo[0], tid);
            }
            acc0 = wmma3(ahi, alo, f1.h0, f1.l0, acc0);
            acc1 = wmma3(ahi, alo, f1.h1, f1.l1, acc1);
            acc2 = wmma3(ahi, alo, f1.h2, f1.l2, acc2);
            __syncthreads();
        }
    }
}

// ---- One-shot weight prep: W (Ktot x 3H) f32 -> fragment-major bf16 hi/lo.
__global__ __launch_bounds__(256)
void prep_weights(const float* __restrict__ W, __bf16* __restrict__ Fhi,
                  __bf16* __restrict__ Flo, int Ktot) {
    const int KT = Ktot / 32;
    const int total = 64 * 3 * KT * 32;
    int idx = blockIdx.x * blockDim.x + threadIdx.x;
    if (idx >= total) return;
    const int lane = idx & 31;
    const int rest = idx >> 5;
    const int kt = rest % KT;
    const int g  = (rest / KT) % 3;
    const int jb = (rest / KT) / 3;
    const int n = lane & 15, half = lane >> 4;
    PackU hi, lo;
    for (int e = 0; e < 16; ++e) {
        int kk = (e < 8) ? (half * 8 + e) : (16 + half * 8 + (e - 8));
        float v = W[(size_t)(kt * 32 + kk) * G3 + g * H_DIM + jb * 16 + n];
        __bf16 h = (__bf16)v;
        hi.h[e] = h;
        lo.h[e] = (__bf16)(v - (float)h);
    }
    size_t base = (size_t)idx * 16;
    *reinterpret_cast<uint4*>(Fhi + base)     = hi.q[0];
    *reinterpret_cast<uint4*>(Fhi + base + 8) = hi.q[1];
    *reinterpret_cast<uint4*>(Flo + base)     = lo.q[0];
    *reinterpret_cast<uint4*>(Flo + base + 8) = lo.q[1];
}

// ---- One recurrence step. grid = H/16 WGs, block = 128 (4 waves).
__global__ __launch_bounds__(128)
void mgruf_step(const float* __restrict__ x_t,       // (B, I)
                const __bf16* __restrict__ Wxf_hi, const __bf16* __restrict__ Wxf_lo,
                const __bf16* __restrict__ Whf_hi, const __bf16* __restrict__ Whf_lo,
                const float* __restrict__ bias,      // (3H)
                const float* __restrict__ mem_para,  // (H)
                float* __restrict__ ring,            // (K, B, H) rotating
                float* __restrict__ h_out_t,         // (B, H) slice of h_all
                int t) {
    __shared__ __bf16 As_hi[2][64][32];   // ping-pong A tile (batch x k)
    __shared__ __bf16 As_lo[2][64][32];

    const int tid  = threadIdx.x;
    const int wave = tid >> 5;
    const int lane = tid & 31;
    const int half = lane >> 4;
    const int lr   = lane & 15;
    const int jb   = blockIdx.x;
    const int j0   = jb * 16;           // h-column base
    const int m0   = wave * 16;         // batch-row base of this wave

    v8f acc_z = {}, acc_r = {}, acc_hn = {}, acc_xn = {};

    const float* h_prev = ring + (size_t)((t + 7) & 7) * (B_DIM * H_DIM);

    // Phase 1: gh = h_prev @ Wh  (z, r, hn accumulators)
    gemm_phase(h_prev, H_DIM, Whf_hi, Whf_lo, H_DIM / 32,
               jb, tid, lane, m0, lr, half, As_hi, As_lo, acc_z, acc_r, acc_hn);
    // Phase 2: gx = x_t @ Wx  (z, r, xn accumulators)
    gemm_phase(x_t, I_DIM, Wxf_hi, Wxf_lo, I_DIM / 32,
               jb, tid, lane, m0, lr, half, As_hi, As_lo, acc_z, acc_r, acc_xn);

    // ---------- Epilogue: gates + fractional memory + state update ----------
    // C/D layout: lane lr -> column N, VGPR r -> M = r + 8*half.
    const int h = j0 + lr;
    const float alpha = 0.5f * sigmf(mem_para[h]);
    float wc[K_DIM];   // wc[0] = coeff of oldest state
    {
        float c = 1.0f;
        for (int i = 0; i < K_DIM; ++i) {
            c *= ((float)i + alpha) / ((float)i + 1.0f);
            wc[K_DIM - 1 - i] = c;
        }
    }
    const float bz = bias[h], br = bias[H_DIM + h], bn = bias[2 * H_DIM + h];
    float* ring_w = ring + (size_t)(t & 7) * (B_DIM * H_DIM);
    for (int r = 0; r < 8; ++r) {
        const int m = m0 + r + half * 8;                 // batch row (exclusive owner)
        float z  = sigmf(acc_z[r] + bz);
        float rg = sigmf(acc_r[r] + br);
        float n  = tanhf(acc_xn[r] + bn + rg * acc_hn[r]);
        float mem = 0.0f;
        for (int i = 0; i < K_DIM; ++i) {                // logical i=0 oldest -> slot (t+i)&7
            const float* st = ring + (size_t)((t + i) & 7) * (B_DIM * H_DIM);
            mem += wc[i] * st[m * H_DIM + h];
        }
        float hn = (1.0f - z) * n + z * mem;
        ring_w[m * H_DIM + h]  = hn;                     // reads above precede this write
        h_out_t[m * H_DIM + h] = hn;
    }
}

// ---- Final projection: out(T*B, O) = h_all(T*B, H) @ Wo(H, O) + bo. Parallel.
__global__ __launch_bounds__(128)
void mgruf_out(const float* __restrict__ h_all,
               const float* __restrict__ Wo,
               const float* __restrict__ bo,
               float* __restrict__ out) {
    __shared__ __bf16 As_hi[64][32], As_lo[64][32];
    __shared__ __bf16 Bs_hi[64][32], Bs_lo[64][32];   // [n][k]
    const int tid = threadIdx.x, wave = tid >> 5, lane = tid & 31;
    const int half = lane >> 4, lr = lane & 15;
    const int r0 = blockIdx.x * 64;   // flat (t*B) row base
    const int n0 = blockIdx.y * 64;   // output-column base
    v8f acc[4] = {{}, {}, {}, {}};

    for (int k0 = 0; k0 < H_DIM; k0 += 32) {
        for (int idx = tid; idx < 64 * 8; idx += 128) {   // float4-grained A staging
            int row = idx >> 3, kq = (idx & 7) * 4;
            float4 v = *reinterpret_cast<const float4*>(&h_all[(size_t)(r0 + row) * H_DIM + k0 + kq]);
            union { __bf16 h[4]; uint2 u; } ph, pl;
            ph.h[0] = (__bf16)v.x; ph.h[1] = (__bf16)v.y;
            ph.h[2] = (__bf16)v.z; ph.h[3] = (__bf16)v.w;
            pl.h[0] = (__bf16)(v.x - (float)ph.h[0]);
            pl.h[1] = (__bf16)(v.y - (float)ph.h[1]);
            pl.h[2] = (__bf16)(v.z - (float)ph.h[2]);
            pl.h[3] = (__bf16)(v.w - (float)ph.h[3]);
            *reinterpret_cast<uint2*>(&As_hi[row][kq]) = ph.u;
            *reinterpret_cast<uint2*>(&As_lo[row][kq]) = pl.u;
        }
        for (int idx = tid; idx < 64 * 32; idx += 128) {  // B transpose staging
            int n = idx & 63, kk = idx >> 6;
            float v = Wo[(size_t)(k0 + kk) * O_DIM + n0 + n];
            __bf16 hi = (__bf16)v;
            Bs_hi[n][kk] = hi;
            Bs_lo[n][kk] = (__bf16)(v - (float)hi);
        }
        __syncthreads();
        v16bf ahi = load_frag(&As_hi[wave * 16 + lr][0], half);
        v16bf alo = load_frag(&As_lo[wave * 16 + lr][0], half);
        for (int nt = 0; nt < 4; ++nt) {
            v16bf bhi = load_frag(&Bs_hi[nt * 16 + lr][0], half);
            v16bf blo = load_frag(&Bs_lo[nt * 16 + lr][0], half);
            acc[nt] = wmma3(ahi, alo, bhi, blo, acc[nt]);
        }
        __syncthreads();
    }
    for (int nt = 0; nt < 4; ++nt) {
        const int col = n0 + nt * 16 + lr;
        const float bb = bo[col];
        for (int r = 0; r < 8; ++r) {
            const int row = r0 + wave * 16 + r + half * 8;
            out[(size_t)row * O_DIM + col] = acc[nt][r] + bb;
        }
    }
}

extern "C" void kernel_launch(void* const* d_in, const int* in_sizes, int n_in,
                              void* d_out, int out_size, void* d_ws, size_t ws_size,
                              hipStream_t stream) {
    (void)in_sizes; (void)n_in; (void)out_size; (void)ws_size;
    const float* inputs   = (const float*)d_in[0];  // (T,B,I)
    const float* hidden0  = (const float*)d_in[1];  // (K,B,H)
    const float* Wx       = (const float*)d_in[2];  // (I,3H)
    const float* Wh       = (const float*)d_in[3];  // (H,3H)
    const float* bias     = (const float*)d_in[4];  // (3H)
    const float* Wo       = (const float*)d_in[5];  // (H,O)
    const float* bo       = (const float*)d_in[6];  // (O)
    const float* mem_para = (const float*)d_in[7];  // (1,H)
    float* out = (float*)d_out;

    // Workspace layout: ring 2MB | h_all 64MB | Whf hi/lo 12.6MB | Wxf hi/lo 6.3MB
    char* ws = (char*)d_ws;
    float*  ring   = (float*)ws;   ws += sizeof(float) * K_DIM * B_DIM * H_DIM;
    float*  h_all  = (float*)ws;   ws += sizeof(float) * (size_t)T_DIM * B_DIM * H_DIM;
    __bf16* Whf_hi = (__bf16*)ws;  ws += sizeof(__bf16) * (size_t)H_DIM * G3;
    __bf16* Whf_lo = (__bf16*)ws;  ws += sizeof(__bf16) * (size_t)H_DIM * G3;
    __bf16* Wxf_hi = (__bf16*)ws;  ws += sizeof(__bf16) * (size_t)I_DIM * G3;
    __bf16* Wxf_lo = (__bf16*)ws;  ws += sizeof(__bf16) * (size_t)I_DIM * G3;

    // Re-initialize ring from inputs every call (deterministic).
    hipMemcpyAsync(ring, hidden0, sizeof(float) * K_DIM * B_DIM * H_DIM,
                   hipMemcpyDeviceToDevice, stream);

    // One-shot weight fragment prep (idempotent, re-run per call).
    {
        int totH = 64 * 3 * (H_DIM / 32) * 32;
        int totX = 64 * 3 * (I_DIM / 32) * 32;
        prep_weights<<<(totH + 255) / 256, 256, 0, stream>>>(Wh, Whf_hi, Whf_lo, H_DIM);
        prep_weights<<<(totX + 255) / 256, 256, 0, stream>>>(Wx, Wxf_hi, Wxf_lo, I_DIM);
    }

    for (int t = 0; t < T_DIM; ++t) {
        mgruf_step<<<dim3(H_DIM / 16), dim3(128), 0, stream>>>(
            inputs + (size_t)t * B_DIM * I_DIM,
            Wxf_hi, Wxf_lo, Whf_hi, Whf_lo, bias, mem_para,
            ring, h_all + (size_t)t * B_DIM * H_DIM, t);
    }

    mgruf_out<<<dim3((T_DIM * B_DIM) / 64, O_DIM / 64), dim3(128), 0, stream>>>(
        h_all, Wo, bo, out);

    // H_fin: after 256 steps, logical order i maps to ring slot i ((256+i)&7 == i).
    hipMemcpyAsync(out + (size_t)T_DIM * B_DIM * O_DIM, ring,
                   sizeof(float) * K_DIM * B_DIM * H_DIM,
                   hipMemcpyDeviceToDevice, stream);
}
